// SparseAttLayer_4776003633306
// MI455X (gfx1250) — matrix operand order repaired and checked by reading, not compiled, not run on hardware
//
#include <hip/hip_runtime.h>
#include <stdint.h>

// SparseAttLayer for MI455X (gfx1250, wave32, WMMA, async global->LDS).
//
//   att[b,k,o,f] = (x[b] @ (bilinear_w[k] @ query[k]^T) * dk^-0.5)[f,o]
//   out[b,k,o,f] = entmax_1.5(att, axis=f) * values[k,o,f]
//
// Stage 1: Mt[k][o][e] = scale * sum_y bw[k,e,y]*q[k,o,y]   (f16, 128 KB in ws)
// Stage 2: per (b,k) block:
//   - async-copy Mt[k] (16 KB) into LDS (global_load_async_to_lds_b128,
//     s_wait_asynccnt) -- one copy per block instead of 16 redundant
//     global B-fragment loads per wave
//   - C[f,o] = x_f16[b](128x128) @ M[k](128x64) via v_wmma_f32_16x16x32_f16
//   - stage C into LDS as att[o][f], entmax bisection register-resident with
//     wave32 shfl_xor butterflies, multiply by values, write out.
//
// Bisection iterations: reference uses 50, but in f32 tau_lo stops changing
// once dm < 0.5*ulp(tau_lo); at iter 32 dm ~= 2e-10, so iters 33..50 are exact
// f32 no-ops for any |tau| > ~1e-3 and perturb p by < 4e-10 otherwise.

typedef __attribute__((ext_vector_type(16))) _Float16 v16h;
typedef __attribute__((ext_vector_type(8)))  float    v8f;

#define NEMB   128
#define NFIELD 128
#define NHID   64
#define NHEAD  8
#define DKK    64
#define LDA    132   // att[o][f] LDS row stride, floats (128+4: breaks bank alias)
#define LDB    144   // Mlds[o][e] LDS row stride, halves (288 B: 32B-aligned rows,
                     // bank stride 72%64=8 -> worst 2-way conflict on b128 reads)
#define N_ITER 32    // f32-converged equivalent of reference's 50 (see header)

// ---------------------------------------------------------------------------
// Stage 1: Mt[k][o][e] = 0.125f * dot(bilinear_w[k][e][:], query[k][o][:])
// ---------------------------------------------------------------------------
__global__ void precompute_Mt(const float* __restrict__ bw,
                              const float* __restrict__ q,
                              _Float16* __restrict__ Mt) {
  const int k = blockIdx.x;
  const float* bwk = bw + (size_t)k * NEMB * DKK;
  const float* qk  = q  + (size_t)k * NHID * DKK;
  for (int idx = threadIdx.x; idx < NHID * NEMB; idx += blockDim.x) {
    const int o = idx >> 7;          // idx = o*128 + e
    const int e = idx & (NEMB - 1);
    const float* br = bwk + (size_t)e * DKK;
    const float* qr = qk  + (size_t)o * DKK;
    float s = 0.f;
#pragma unroll
    for (int y = 0; y < DKK; ++y) s = fmaf(br[y], qr[y], s);
    Mt[(size_t)k * NHID * NEMB + idx] = (_Float16)(s * 0.125f);  // dk^-0.5 baked in
  }
}

// ---------------------------------------------------------------------------
// Stage 2: async B-stage + GEMM (WMMA f16) + entmax bisection + values mult
// ---------------------------------------------------------------------------
__global__ __launch_bounds__(256)
void sparse_att_kernel(const float* __restrict__ x,
                       const _Float16* __restrict__ Mt,
                       const float* __restrict__ vals,
                       float* __restrict__ out) {
  __shared__ float    att[NHID * LDA];    // att[o][f], 33,792 B
  __shared__ _Float16 Mlds[NHID * LDB];   // Mt[k] padded, 18,432 B

  const int bid  = blockIdx.x;            // bid = b*8 + k
  const int k    = bid & (NHEAD - 1);
  const int tid  = threadIdx.x;
  const int w    = tid >> 5;              // wave 0..7 (wave32)
  const int lane = tid & 31;
  const int g    = lane >> 4;             // lane half-group
  const int m    = lane & 15;             // A row / B,C column within tile

  const _Float16* Mk = Mt + (size_t)k * NHID * NEMB;

  // -- Async-stage Mt[k] (64 rows x 128 halves = 1024 x 16 B) into LDS. ------
  // Each of 256 threads DMAs 4 x b128 chunks; ASYNCcnt fences own wave,
  // __syncthreads() publishes across waves.
#pragma unroll
  for (int i = 0; i < 4; ++i) {
    const int c    = tid + i * 256;       // chunk 0..1023
    const int row  = c >> 4;
    const int col8 = (c & 15) * 8;        // halves
    const _Float16* gsrc = Mk + (size_t)row * NEMB + col8;
    const uint32_t  ldst = (uint32_t)(uintptr_t)(&Mlds[row * LDB + col8]);
    asm volatile("global_load_async_to_lds_b128 %0, %1, off"
                 :: "v"(ldst), "v"(gsrc) : "memory");
  }
  asm volatile("s_wait_asynccnt 0x0" ::: "memory");
  __syncthreads();

  // ---------------- GEMM: C[f,o] = x_f16[b] @ Mlds^T ------------------------
  // Wave w owns f-tile row w: f in [16w, 16w+16), all 4 o-tiles.
  const float* xb = x + ((size_t)(bid >> 3) * NFIELD + (size_t)w * 16 + m) * NEMB;

  v8f acc[4] = {v8f{}, v8f{}, v8f{}, v8f{}};
#pragma unroll
  for (int ec = 0; ec < 4; ++ec) {        // K (=nemb) chunks of 32
    const int e0 = ec * 32;
    // A fragment 16x32 f16: lane holds row f=16w+m; halves j = K e0+16g+j
    // (matches B order). Convert f32->f16.
    const float* pa = xb + e0 + g * 16;
    const float4 q0 = *(const float4*)(pa);
    const float4 q1 = *(const float4*)(pa + 4);
    const float4 q2 = *(const float4*)(pa + 8);
    const float4 q3 = *(const float4*)(pa + 12);
    v16h a;
    a[0]=(_Float16)q0.x;  a[1]=(_Float16)q0.y;  a[2]=(_Float16)q0.z;  a[3]=(_Float16)q0.w;
    a[4]=(_Float16)q1.x;  a[5]=(_Float16)q1.y;  a[6]=(_Float16)q1.z;  a[7]=(_Float16)q1.w;
    a[8]=(_Float16)q2.x;  a[9]=(_Float16)q2.y;  a[10]=(_Float16)q2.z; a[11]=(_Float16)q2.w;
    a[12]=(_Float16)q3.x; a[13]=(_Float16)q3.y; a[14]=(_Float16)q3.z; a[15]=(_Float16)q3.w;
#pragma unroll
    for (int ot = 0; ot < 4; ++ot) {
      // B fragment 32x16 f16: lane = column N=m, halves j = K = 16g+j.
      // 32-B aligned ds_load from the staged (padded) LDS tile.
      const v16h bf = *(const v16h*)(&Mlds[(ot * 16 + m) * LDB + e0 + g * 16]);
      acc[ot] = __builtin_amdgcn_wmma_f32_16x16x32_f16(
          false, a, false, bf, (short)0, acc[ot], false, false);
    }
  }

  // Stage C -> LDS transposed to att[o][f]. C layout: elem r => f = 16w+8g+r,
  // o = 16*ot + m. r=0..7 are f-contiguous => two aligned float4 DS stores.
#pragma unroll
  for (int ot = 0; ot < 4; ++ot) {
    float* dst = &att[(ot * 16 + m) * LDA + w * 16 + g * 8];
    *(float4*)(dst)     = float4{acc[ot][0], acc[ot][1], acc[ot][2], acc[ot][3]};
    *(float4*)(dst + 4) = float4{acc[ot][4], acc[ot][5], acc[ot][6], acc[ot][7]};
  }

  // Warm L2/L0 with this wave's values rows (needed after the ~us bisection).
#pragma unroll
  for (int r = 0; r < 8; ++r)
    __builtin_prefetch(vals + ((size_t)k * NHID + w * 8 + r) * NFIELD + lane * 4, 0, 1);

  __syncthreads();

  // ---------------- entmax_1.5 bisection over f (row length 128) ------------
  // Wave w handles o-rows [8w, 8w+8); 4 concurrent rows for ILP across the
  // serial bisection chain. Lane holds f = lane + 32j, j=0..3 (conflict-free).
  const float DM0 = 0.91161165235168155f;  // 1 - (1/128)^(alpha-1), alpha=1.5
#pragma unroll
  for (int rr = 0; rr < 2; ++rr) {
    const int obase = w * 8 + rr * 4;
    float xs[4][4];
#pragma unroll
    for (int r = 0; r < 4; ++r)
#pragma unroll
      for (int j = 0; j < 4; ++j)
        xs[r][j] = 0.5f * att[(obase + r) * LDA + lane + 32 * j];  // *(alpha-1)

    float tau_lo[4], f_lo[4];
#pragma unroll
    for (int r = 0; r < 4; ++r) {
      float mx = fmaxf(fmaxf(xs[r][0], xs[r][1]), fmaxf(xs[r][2], xs[r][3]));
#pragma unroll
      for (int sh = 16; sh >= 1; sh >>= 1) mx = fmaxf(mx, __shfl_xor(mx, sh, 32));
      tau_lo[r] = mx - 1.0f;
      float s = 0.f;
#pragma unroll
      for (int j = 0; j < 4; ++j) {
        float t = fmaxf(xs[r][j] - tau_lo[r], 0.f);
        s = fmaf(t, t, s);
      }
#pragma unroll
      for (int sh = 16; sh >= 1; sh >>= 1) s += __shfl_xor(s, sh, 32);
      f_lo[r] = s - 1.0f;
    }

    float dm = DM0;
    for (int it = 0; it < N_ITER; ++it) {
      dm *= 0.5f;
      float sm[4];
#pragma unroll
      for (int r = 0; r < 4; ++r) {
        const float tau = tau_lo[r] + dm;
        float s = 0.f;
#pragma unroll
        for (int j = 0; j < 4; ++j) {
          float t = fmaxf(xs[r][j] - tau, 0.f);
          s = fmaf(t, t, s);
        }
        sm[r] = s;
      }
#pragma unroll
      for (int sh = 16; sh >= 1; sh >>= 1)
#pragma unroll
        for (int r = 0; r < 4; ++r) sm[r] += __shfl_xor(sm[r], sh, 32);
#pragma unroll
      for (int r = 0; r < 4; ++r)
        tau_lo[r] = ((sm[r] - 1.0f) * f_lo[r] >= 0.f) ? (tau_lo[r] + dm) : tau_lo[r];
    }

    // Final probabilities, normalize, multiply by values, store coalesced.
#pragma unroll
    for (int r = 0; r < 4; ++r) {
      const int o = obase + r;
      const float tauf = tau_lo[r] + dm;
      float p[4], S = 0.f;
#pragma unroll
      for (int j = 0; j < 4; ++j) {
        float t = fmaxf(xs[r][j] - tauf, 0.f);
        p[j] = t * t;
        S += p[j];
      }
#pragma unroll
      for (int sh = 16; sh >= 1; sh >>= 1) S += __shfl_xor(S, sh, 32);
      const float inv = 1.0f / S;
      const float* vrow = vals + ((size_t)k * NHID + o) * NFIELD;
      float*       orow = out  + ((size_t)bid * NHID + o) * NFIELD;
#pragma unroll
      for (int j = 0; j < 4; ++j)
        orow[lane + 32 * j] = (p[j] * inv) * vrow[lane + 32 * j];
    }
  }
}

// ---------------------------------------------------------------------------
extern "C" void kernel_launch(void* const* d_in, const int* in_sizes, int n_in,
                              void* d_out, int out_size, void* d_ws, size_t ws_size,
                              hipStream_t stream) {
  (void)n_in; (void)out_size; (void)ws_size;
  const float* x  = (const float*)d_in[0];   // [bsz, 128, 128]
  const float* bw = (const float*)d_in[1];   // [8, 128, 64]
  const float* q  = (const float*)d_in[2];   // [8, 64, 64]
  const float* v  = (const float*)d_in[3];   // [8, 64, 128]
  float* out = (float*)d_out;                // [bsz, 8, 64, 128]
  _Float16* Mt = (_Float16*)d_ws;            // [8, 64, 128] f16 = 128 KB

  const int bsz = in_sizes[0] / (NFIELD * NEMB);

  precompute_Mt<<<NHEAD, 256, 0, stream>>>(bw, q, Mt);
  sparse_att_kernel<<<bsz * NHEAD, 256, 0, stream>>>(x, Mt, v, out);
}
